// RNN_47476568490335
// MI455X (gfx1250) — matrix operand order repaired and checked by reading, not compiled, not run on hardware
//
#include <hip/hip_runtime.h>
#include <math.h>

// CDNA5 wave32 WMMA vector types
typedef _Float16 v16h __attribute__((ext_vector_type(16)));
typedef _Float16 v8h  __attribute__((ext_vector_type(8)));
typedef _Float16 v4h  __attribute__((ext_vector_type(4)));
typedef float    v8f  __attribute__((ext_vector_type(8)));

#define Hn 256
#define Vv 64
#define Bn 2048
#define Tn 512
#define KTOT 320          // H + V
#define NKC  10           // K chunks of 32
#define TH_STRIDE 328     // theta LDS row stride in halves (16B-aligned, bank-spread)
#define ST_STRIDE 328     // state LDS row stride in halves
#define Z_STRIDE  264     // z LDS row stride in floats (16B-aligned)
#define BT 16             // batch rows per block

#define SMEM_BYTES (Hn*TH_STRIDE*2 + BT*ST_STRIDE*2 + BT*Z_STRIDE*4)

__device__ __forceinline__ v16h load_afrag(const _Float16* arow, int kc) {
    v8h alo = *(const v8h*)(arow + kc);
    v8h ahi = *(const v8h*)(arow + kc + 16);
    return __builtin_shufflevector(alo, ahi,
            0,1,2,3,4,5,6,7,8,9,10,11,12,13,14,15);
}

__global__ __launch_bounds__(256, 1)
void rnn_scan_wmma(const float* __restrict__ x,      // [B,T,V] f32
                   const float* __restrict__ theta,  // [H, H+V] f32
                   const float* __restrict__ bias,   // [1] f32  (cancels under layernorm)
                   const float* __restrict__ tdot,   // [1,H] f32
                   float* __restrict__ out)          // [B] f32
{
    extern __shared__ char smem[];
    _Float16* th  = (_Float16*)smem;                                   // [256][328] f16
    _Float16* st  = (_Float16*)(smem + Hn*TH_STRIDE*2);                // [16][328]  f16 ([state|x])
    float*    zbf = (float*)   (smem + Hn*TH_STRIDE*2 + BT*ST_STRIDE*2); // [16][264] f32

    const int tid   = threadIdx.x;
    const int lane  = tid & 31;
    const int wave  = tid >> 5;        // 0..7
    const int half  = lane >> 4;       // 0/1
    const int ln16  = lane & 15;
    const int bbase = blockIdx.x * BT;
    (void)bias;  // z + b - mean(z + b) == z - mean(z): bias cancels exactly

    // ---- one-time: theta f32 -> f16 into LDS, zero state ----
    for (int i = tid; i < Hn * KTOT; i += 256) {
        int r = i / KTOT, c = i - r * KTOT;
        th[r * TH_STRIDE + c] = (_Float16)theta[i];
    }
    for (int i = tid; i < BT * ST_STRIDE; i += 256)
        st[i] = (_Float16)0.0f;

    // per-thread token slice: 4 consecutive floats of x[bbase+xrow, t, xcol..xcol+3]
    const int xe   = tid * 4;          // 0..1023 over 16x64 tile
    const int xrow = xe >> 6;
    const int xcol = xe & 63;
    const float* xptr = x + ((size_t)(bbase + xrow) * Tn) * Vv + xcol;

    const int nb   = wave * 32;        // this wave's two 16-wide N tiles
    const int srow = tid >> 4;         // stats/normalize row   (0..15)
    const int scol = (tid & 15) * 16;  // stats/normalize col base

    const _Float16* arow  = st + ln16 * ST_STRIDE + half * 8;
    const _Float16* brow0 = th + (nb + ln16) * TH_STRIDE + half * 8;
    const _Float16* brow1 = brow0 + 16 * TH_STRIDE;
    _Float16* xdst = st + xrow * ST_STRIDE + Hn + xcol;

    // ---- hoist theta B-fragments into registers: loop-invariant over all T steps.
    // 20 x v16h = 160 VGPRs/wave; single workgroup per WGP (LDS-bound), budget OK.
    __syncthreads();
    v16h bfr[2 * NKC];
#pragma unroll
    for (int kk = 0; kk < NKC; ++kk) {
        int kc = kk * 32;
        v8h bl0 = *(const v8h*)(brow0 + kc);
        v8h bh0 = *(const v8h*)(brow0 + kc + 16);
        bfr[2*kk]   = __builtin_shufflevector(bl0, bh0,
                        0,1,2,3,4,5,6,7,8,9,10,11,12,13,14,15);
        v8h bl1 = *(const v8h*)(brow1 + kc);
        v8h bh1 = *(const v8h*)(brow1 + kc + 16);
        bfr[2*kk+1] = __builtin_shufflevector(bl1, bh1,
                        0,1,2,3,4,5,6,7,8,9,10,11,12,13,14,15);
    }

    // ---- prologue: stage token 0, preload token 1 into registers ----
    {
        float4 x0 = *(const float4*)(xptr);
        v4h hx;
        hx[0] = (_Float16)x0.x; hx[1] = (_Float16)x0.y;
        hx[2] = (_Float16)x0.z; hx[3] = (_Float16)x0.w;
        *(v4h*)(xdst) = hx;
    }
    float4 xreg = *(const float4*)(xptr + (size_t)Vv);  // token t=1
    __syncthreads();

    for (int t = 0; t < Tn; ++t) {
        // ---- z[16, nb..nb+32) = st[16,320] @ theta[nb..nb+32, 320]^T ----
        if (t + 3 < Tn) __builtin_prefetch(xptr + (size_t)(t + 3) * Vv, 0, 3);

        v8f acc0 = {}; v8f acc1 = {};
        v16h a_cur = load_afrag(arow, 0);
#pragma unroll
        for (int kk = 0; kk < NKC; ++kk) {
            v16h a_nxt;
            if (kk + 1 < NKC) a_nxt = load_afrag(arow, (kk + 1) * 32);
            acc0 = __builtin_amdgcn_wmma_f32_16x16x32_f16(
                       false, a_cur, false, bfr[2*kk],   (short)0, acc0, false, false);
            acc1 = __builtin_amdgcn_wmma_f32_16x16x32_f16(
                       false, a_cur, false, bfr[2*kk+1], (short)0, acc1, false, false);
            a_cur = a_nxt;
        }

        // spill z tiles: lane holds D[m = r + 8*half][n = ln16]
#pragma unroll
        for (int r = 0; r < 8; ++r) {
            int m = r + half * 8;
            zbf[m * Z_STRIDE + nb + ln16]      = acc0[r];
            zbf[m * Z_STRIDE + nb + 16 + ln16] = acc1[r];
        }
        __syncthreads();   // barrier A: all WMMA reads + z writes done

        // ---- layernorm (ddof=1) + relu: thread owns z[srow, scol..scol+15] ----
        float zv[16];
        float s = 0.0f, q = 0.0f;
#pragma unroll
        for (int j = 0; j < 16; j += 4) {
            float4 zz = *(const float4*)(zbf + srow * Z_STRIDE + scol + j);
            zv[j+0] = zz.x; zv[j+1] = zz.y; zv[j+2] = zz.z; zv[j+3] = zz.w;
            s += zz.x + zz.y + zz.z + zz.w;
            q += zz.x*zz.x + zz.y*zz.y + zz.z*zz.z + zz.w*zz.w;
        }
#pragma unroll
        for (int m = 1; m <= 8; m <<= 1) {   // butterfly over the 16-lane row group
            s += __shfl_xor(s, m);
            q += __shfl_xor(q, m);
        }
        float mean = s * (1.0f / Hn);
        float var  = (q - s * mean) * (1.0f / (Hn - 1));
        float rstd = rsqrtf(fmaxf(var, 1e-20f));

        v8h h0, h1;
#pragma unroll
        for (int j = 0; j < 16; ++j) {
            float zn = (zv[j] - mean) * rstd;
            zn = fmaxf(zn, 0.0f);
            if (j < 8) h0[j]     = (_Float16)zn;
            else       h1[j - 8] = (_Float16)zn;
        }
        *(v8h*)(st + srow * ST_STRIDE + scol)     = h0;
        *(v8h*)(st + srow * ST_STRIDE + scol + 8) = h1;

        // stage token t+1 (held in registers) and start load of token t+2
        if (t + 1 < Tn) {
            v4h hx;
            hx[0] = (_Float16)xreg.x; hx[1] = (_Float16)xreg.y;
            hx[2] = (_Float16)xreg.z; hx[3] = (_Float16)xreg.w;
            *(v4h*)(xdst) = hx;
            if (t + 2 < Tn) xreg = *(const float4*)(xptr + (size_t)(t + 2) * Vv);
        }
        __syncthreads();   // barrier B: state + token staged for step t+1
    }

    // logits = sum(state * theta_dot) ; sigmoid
    float d = 0.0f;
#pragma unroll
    for (int j = 0; j < 16; ++j)
        d += (float)st[srow * ST_STRIDE + scol + j] * tdot[scol + j];
#pragma unroll
    for (int m = 1; m <= 8; m <<= 1)
        d += __shfl_xor(d, m);
    if ((tid & 15) == 0)
        out[bbase + srow] = 1.0f / (1.0f + __expf(-d));
}

extern "C" void kernel_launch(void* const* d_in, const int* in_sizes, int n_in,
                              void* d_out, int out_size, void* d_ws, size_t ws_size,
                              hipStream_t stream) {
    (void)in_sizes; (void)n_in; (void)d_ws; (void)ws_size; (void)out_size;
    const float* x     = (const float*)d_in[0];  // [B,T,V]
    const float* theta = (const float*)d_in[1];  // [H,H+V]
    const float* bias  = (const float*)d_in[2];  // [1]
    const float* tdot  = (const float*)d_in[3];  // [1,H]
    float* out = (float*)d_out;                  // [B]

    // ~191 KB dynamic LDS (fits CDNA5's 320 KB/WGP); opt in above default cap.
    hipFuncSetAttribute((const void*)rnn_scan_wmma,
                        hipFuncAttributeMaxDynamicSharedMemorySize, SMEM_BYTES);

    dim3 grid(Bn / BT);   // 128 blocks
    dim3 block(256);      // 8 wave32
    rnn_scan_wmma<<<grid, block, SMEM_BYTES, stream>>>(x, theta, bias, tdot, out);
}
